// STDPLayer_81621558493185
// MI455X (gfx1250) — compile-verified, matrix operand order checked by reading, not run on hardware
//
#include <hip/hip_runtime.h>

// ---------------------------------------------------------------------------
// STDP update for MI455X (gfx1250), wave32 + WMMA + async-to-LDS staging.
//   pre_spikes  [T=200, B=64, IN=2500]  f32
//   post_spikes [T=200, B=64, OUT=128]  f32
//   weights     [OUT=128, IN=2500]      f32
//   out         [OUT=128, IN=2500]      f32
//
// Fused formulation (A_PLUS == A_MINUS == 0.01):
//   delta[i,j] = (0.01/(T*B)) * sum_k ( pre_trace[k,i]*post[k,j]
//                                     - pre[k,i]*post_trace[k,j] )
// One f16 GEMM with K' = 2*T*B = 25600:
//   A' = [pre_trace ; pre_spikes]   (K' x Mpad), Mpad = 2560 (zero-padded)
//   B' = [post      ; -post_trace]  (K' x 128)
// ---------------------------------------------------------------------------

typedef _Float16 v16h __attribute__((ext_vector_type(16)));
typedef float    v8f  __attribute__((ext_vector_type(8)));

#define T_STEPS 200
#define BATCH   64
#define IN_N    2500
#define OUT_N   128
#define MPAD    2560            // IN rounded up to 128
#define KTB     (T_STEPS*BATCH) // 12800
#define K2      (2*KTB)         // 25600
#define KT32    (K2/32)         // 800  (# of 32-wide K tiles)
#define MT16    (MPAD/16)       // 160  (# of 16-wide M tiles)
#define NT16    (OUT_N/16)      // 8
#define NKCHUNK 16
#define KSTEPS  (KT32/NKCHUNK)  // 50 K32-steps per chunk
#define DECAY   0.95122942450071400910f   // exp(-1/20)
#define SCALE   7.8125e-7f                // 0.01 / (T*B)

// ---------------------------------------------------------------------------
// CDNA5 async global->LDS copy (ASYNCcnt path), with safe fallback.
// Builtin signature (confirmed by compiler diagnostic):
//   (b128 AS1*, b128 AS3*, imm offset, imm cpol)
// ---------------------------------------------------------------------------
#if defined(__has_builtin)
#  if __has_builtin(__builtin_amdgcn_global_load_async_to_lds_b128)
#    define HAVE_ASYNC_LDS 1
#  endif
#endif
#ifndef HAVE_ASYNC_LDS
#  define HAVE_ASYNC_LDS 0
#endif

#if HAVE_ASYNC_LDS
typedef int b128_t __attribute__((vector_size(4 * sizeof(int))));
typedef __attribute__((address_space(1))) b128_t* glob_b128;
typedef __attribute__((address_space(3))) b128_t* lds_b128;
// Flat->LDS address is addr[31:0] (ISA 10.2 aperture table); flat->global is
// the identity, so integer-truncation casts build valid AS pointers.
__device__ __forceinline__ glob_b128 to_glob(const void* p) {
  return (glob_b128)(unsigned long long)(uintptr_t)p;
}
__device__ __forceinline__ lds_b128 to_lds(void* p) {
  return (lds_b128)(unsigned int)(uintptr_t)p;
}
// copy 32 bytes per lane (one v16h fragment slice) = 2x b128
__device__ __forceinline__ void async_copy_frag(const _Float16* g, _Float16* l) {
  __builtin_amdgcn_global_load_async_to_lds_b128(to_glob(g),     to_lds(l),     0, 0);
  __builtin_amdgcn_global_load_async_to_lds_b128(to_glob(g + 8), to_lds(l + 8), 0, 0);
}
__device__ __forceinline__ void wait_async0() {
#  if __has_builtin(__builtin_amdgcn_s_wait_asynccnt)
  __builtin_amdgcn_s_wait_asynccnt(0);
#  else
  asm volatile("s_wait_asynccnt 0x0" ::: "memory");
#  endif
}
#endif

// Position of element (kk in [0,32), mm in [0,16)) inside a 1KB fragment
// block, in half-units. Matches ISA 7.12.2 "16-bit A-Matrix 16x32":
//   lanes 0-15 : M=mm, elems 0..7 -> K 0..7,  elems 8..15 -> K 16..23
//   lanes 16-31: M=mm, elems 0..7 -> K 8..15, elems 8..15 -> K 24..31
__device__ __forceinline__ int frag_off(int kk, int mm) {
  int hi   = (kk >> 3) & 1;                 // bit3 of kk selects lane half
  int lane = mm + (hi << 4);
  int elem = (kk & 7) | ((kk & 16) >> 1);   // +8 when kk >= 16 (per half)
  return lane * 16 + elem;                  // halfs within 512-half block
}

// ---------------------------------------------------------------------------
// Kernel 1: leaky trace over pre_spikes + pack f16 fragment blocks for A'.
// ---------------------------------------------------------------------------
__global__ __launch_bounds__(256)
void stdp_pack_pre(const float* __restrict__ pre, _Float16* __restrict__ ablk) {
  int tid = blockIdx.x * blockDim.x + threadIdx.x;
  if (tid >= BATCH * MPAD) return;
  int b = tid / MPAD;
  int m = tid - b * MPAD;
  bool valid = (m < IN_N);

  // k = t*BATCH + b  ->  kk = b & 31 (t*64 == 0 mod 32), kt = 2t + (b>>5)
  int  foff = frag_off(b & 31, m & 15);
  long long idx = ((long long)((b >> 5) * MT16 + (m >> 4))) * 512 + foff;
  const long long step      = 2LL * MT16 * 512;      // kt advances by 2 per t
  const long long spike_off = 400LL * MT16 * 512;    // + KTB rows (KTB/32=400)

  float trace = 0.0f;
  long long src = (long long)b * IN_N + m;
  for (int t = 0; t < T_STEPS; ++t) {
    float s = valid ? pre[src] : 0.0f;
    trace = trace * DECAY + s;
    ablk[idx]             = (_Float16)trace;   // top half of A'  (pre_trace)
    ablk[idx + spike_off] = (_Float16)s;       // bottom half     (pre spikes)
    idx += step;
    src += (long long)BATCH * IN_N;
  }
}

// ---------------------------------------------------------------------------
// Kernel 2: leaky trace over post_spikes + pack f16 fragment blocks for B'.
// ---------------------------------------------------------------------------
__global__ __launch_bounds__(256)
void stdp_pack_post(const float* __restrict__ post, _Float16* __restrict__ bblk) {
  int tid = blockIdx.x * blockDim.x + threadIdx.x;
  if (tid >= BATCH * OUT_N) return;
  int b = tid >> 7;
  int j = tid & 127;

  int  foff = frag_off(b & 31, j & 15);
  long long idx = ((long long)((b >> 5) * NT16 + (j >> 4))) * 512 + foff;
  const long long step      = 2LL * NT16 * 512;
  const long long trace_off = 400LL * NT16 * 512;

  float trace = 0.0f;
  long long src = (long long)b * OUT_N + j;
  for (int t = 0; t < T_STEPS; ++t) {
    float s = post[src];
    trace = trace * DECAY + s;
    bblk[idx]             = (_Float16)s;        // top half of B'
    bblk[idx + trace_off] = (_Float16)(-trace); // bottom half (negated)
    idx += step;
    src += (long long)BATCH * OUT_N;
  }
}

// ---------------------------------------------------------------------------
// Kernel 3: split-K WMMA GEMM. grid = (MPAD/128, NKCHUNK), 512 thr = 16 waves.
// Wave w = (mhalf, nt): waves 0-7 do m-tiles 0-3 of n-tile w, waves 8-15 do
// m-tiles 4-7. Per-wave: 4 accumulators + 4 A-frags -> low VGPR pressure.
// A fragments staged into a double-buffered LDS tile by waves 0-7 using
// GLOBAL_LOAD_ASYNC_TO_LDS_B128 (ASYNCcnt): while step s computes its WMMAs,
// the async engine fills step s+1. One barrier per K-step.
// Partials stored per chunk (plain stores -> deterministic).
// ---------------------------------------------------------------------------
__global__ __launch_bounds__(512)
void stdp_gemm(const _Float16* __restrict__ ablk,
               const _Float16* __restrict__ bblk,
               float* __restrict__ cpart) {
  __shared__ __align__(64) _Float16 lds_a[2][8 * 512];  // 2 x 8KB

  const int lane   = threadIdx.x & 31;
  const int wave   = threadIdx.x >> 5;   // 0..15
  const int nt     = wave & 7;           // n-tile index
  const int mhalf  = wave >> 3;          // 0: m-tiles 0-3, 1: m-tiles 4-7
  const bool stager = (wave < 8);        // waves 0-7 stage fragment 'wave'
  const int mblock = blockIdx.x;         // 0..19  (8 m-tiles each)
  const int chunk  = blockIdx.y;         // 0..15
  const int kt0    = chunk * KSTEPS;

  v8f acc[4] = {};

  const long long a_step = (long long)MT16 * 512;
  // staging source for this wave's fragment (frag index = wave&7)
  long long a_off = ((long long)kt0 * MT16 + mblock * 8 + (wave & 7)) * 512 + lane * 16;
  long long b_off = ((long long)kt0 * NT16 + nt) * 512 + lane * 16;

  _Float16* slot0 = &lds_a[0][(wave & 7) * 512 + lane * 16];
  _Float16* slot1 = &lds_a[1][(wave & 7) * 512 + lane * 16];

  // ---- prologue: stage step 0 into buffer 0 ----
#if HAVE_ASYNC_LDS
  if (stager) async_copy_frag(ablk + a_off, slot0);
  wait_async0();
#else
  if (stager) *(v16h*)slot0 = *(const v16h*)(ablk + a_off);
#endif
  __syncthreads();

  for (int s = 0; s < KSTEPS; ++s) {
    v16h bfrag = *(const v16h*)(bblk + b_off);
    const _Float16* cur = ((s & 1) ? lds_a[1] : lds_a[0]) + mhalf * 4 * 512 + lane * 16;
    _Float16* nxt = (s & 1) ? slot0 : slot1;
#if !HAVE_ASYNC_LDS
    v16h pend = {};
#endif
    if (s + 1 < KSTEPS) {
#if HAVE_ASYNC_LDS
      if (stager) async_copy_frag(ablk + a_off + a_step, nxt);  // fill other buf
#else
      if (stager) pend = *(const v16h*)(ablk + a_off + a_step);
#endif
      __builtin_prefetch(bblk + b_off + (long long)NT16 * 512, 0, 1);
    }

#pragma unroll
    for (int mt = 0; mt < 4; ++mt) {
      v16h a = *(const v16h*)(cur + mt * 512);
      acc[mt] = __builtin_amdgcn_wmma_f32_16x16x32_f16(
          false, a, false, bfrag, (short)0, acc[mt], false, false);
    }

#if !HAVE_ASYNC_LDS
    if (stager && s + 1 < KSTEPS) *(v16h*)nxt = pend;
#else
    wait_async0();                         // my async writes landed in LDS
#endif
    __syncthreads();                       // everyone's writes visible / WAR
    a_off += a_step;
    b_off += (long long)NT16 * 512;
  }

  // C/D layout (ISA 7.12.2): VGPR r -> M = r (+8 for lanes 16-31), N = lane%16
  float* base = cpart + (long long)chunk * MPAD * OUT_N;
  const int col    = nt * 16 + (lane & 15);
  const int rowadd = (lane >> 4) * 8;
#pragma unroll
  for (int mt = 0; mt < 4; ++mt) {
    int row0 = (mblock * 8 + mhalf * 4 + mt) * 16 + rowadd;
    float* p = base + (long long)row0 * OUT_N + col;
#pragma unroll
    for (int r = 0; r < 8; ++r) p[(long long)r * OUT_N] = acc[mt][r];
  }
}

// ---------------------------------------------------------------------------
// Kernel 4: reduce split-K partials, scale, add weights, clip to [0,1].
// ---------------------------------------------------------------------------
__global__ __launch_bounds__(256)
void stdp_finalize(const float* __restrict__ cpart,
                   const float* __restrict__ w,
                   float* __restrict__ out) {
  int tid = blockIdx.x * blockDim.x + threadIdx.x;
  if (tid >= IN_N * OUT_N) return;
  int i = tid >> 7;        // 0..2499
  int j = tid & 127;       // 0..127
  float s = 0.0f;
#pragma unroll
  for (int c = 0; c < NKCHUNK; ++c)
    s += cpart[(long long)c * MPAD * OUT_N + (long long)i * OUT_N + j];
  long long o = (long long)j * IN_N + i;   // output is [OUT, IN]
  float v = w[o] + SCALE * s;
  out[o] = fminf(fmaxf(v, 0.0f), 1.0f);
}

// ---------------------------------------------------------------------------
extern "C" void kernel_launch(void* const* d_in, const int* in_sizes, int n_in,
                              void* d_out, int out_size, void* d_ws, size_t ws_size,
                              hipStream_t stream) {
  const float* pre  = (const float*)d_in[0];  // [T,B,IN]
  const float* post = (const float*)d_in[1];  // [T,B,OUT]
  const float* wts  = (const float*)d_in[2];  // [OUT,IN]
  float* out = (float*)d_out;

  // Workspace layout (all offsets 1KB-aligned):
  //   cpart : NKCHUNK * MPAD * OUT_N f32   = 20.97 MB
  //   bblk  : KT32 * NT16 * 512 halfs      =  6.55 MB
  //   ablk  : KT32 * MT16 * 512 halfs      = 131.07 MB
  float*    cpart = (float*)d_ws;
  _Float16* bblk  = (_Float16*)(cpart + (long long)NKCHUNK * MPAD * OUT_N);
  _Float16* ablk  = bblk + (long long)KT32 * NT16 * 512;

  stdp_pack_pre <<<(BATCH * MPAD  + 255) / 256, 256, 0, stream>>>(pre,  ablk);
  stdp_pack_post<<<(BATCH * OUT_N + 255) / 256, 256, 0, stream>>>(post, bblk);
  stdp_gemm     <<<dim3(MPAD / 128, NKCHUNK), 512, 0, stream>>>(ablk, bblk, cpart);
  stdp_finalize <<<(IN_N * OUT_N + 255) / 256, 256, 0, stream>>>(cpart, wts, out);
}
